// FINGER_ID_85057532330127
// MI455X (gfx1250) — matrix operand order repaired and checked by reading, not compiled, not run on hardware
//
#include <hip/hip_runtime.h>

#define N_NODES 16384
#define B_MOL   256
#define NPM     64
#define E_EDGES 65536
#define FMAX    32
#define AMAX    16
#define HID     960
#define FGE     512
#define POSE    128
#define NFEAT   5
#define FUSEIN  2560
#define BF      (B_MOL*FMAX)          /* 8192 FG slots */
#define NT16    (HID/16)              /* 60 column tiles of 16 */
#define NCG     (HID/64)              /* 15 column groups of 64 */

typedef __bf16 bf16_t;
typedef bf16_t v16bf __attribute__((ext_vector_type(16)));
typedef float  v8f   __attribute__((ext_vector_type(8)));
typedef unsigned short u16;
typedef unsigned int   u32;

__device__ __forceinline__ u16 f2bf(float f) {
  u32 u = __float_as_uint(f);
  u32 r = u + 0x7FFFu + ((u >> 16) & 1u);   // round-to-nearest-even
  return (u16)(r >> 16);
}

union Frag16 { v16bf v; uint4 q[2]; };

// ---------------- graph preprocessing ----------------

__global__ void k_count(const int* __restrict__ dst, int* __restrict__ deg) {
  int e = blockIdx.x * 256 + threadIdx.x;
  if (e < E_EDGES) atomicAdd(&deg[dst[e]], 1);
}

// single-block exclusive scan of 16384 degree counts -> rowptr; also dinv
__global__ void k_scan(const int* __restrict__ deg, int* __restrict__ rowptr,
                       float* __restrict__ dinv) {
  __shared__ int ssum[1024];
  int t = threadIdx.x;
  int base = t * 16;
  int loc[16]; int s = 0;
  #pragma unroll
  for (int i = 0; i < 16; i++) { loc[i] = deg[base + i]; s += loc[i]; }
  ssum[t] = s; __syncthreads();
  for (int off = 1; off < 1024; off <<= 1) {
    int v = 0;
    if (t >= off) v = ssum[t - off];
    __syncthreads();
    ssum[t] += v;
    __syncthreads();
  }
  int run = (t == 0) ? 0 : ssum[t - 1];
  #pragma unroll
  for (int i = 0; i < 16; i++) {
    rowptr[base + i] = run;
    dinv[base + i] = rsqrtf(1.0f + (float)loc[i]);
    run += loc[i];
  }
  if (t == 1023) rowptr[N_NODES] = run;
}

__global__ void k_scatter(const int* __restrict__ src, const int* __restrict__ dst,
                          const int* __restrict__ rowptr, int* __restrict__ cursor,
                          int* __restrict__ csr) {
  int e = blockIdx.x * 256 + threadIdx.x;
  if (e >= E_EDGES) return;
  int d = dst[e];
  int pos = atomicAdd(&cursor[d], 1);
  csr[rowptr[d] + pos] = src[e];
}

// ---------------- tiny first linear (K=5) ----------------

__global__ void k_lin1(const float* __restrict__ x, const float* __restrict__ W1,
                       float* __restrict__ out) {
  int g = blockIdx.x * 256 + threadIdx.x;
  if (g >= N_NODES * HID) return;
  int n = g / HID, j = g - n * HID;
  const float* xr = x + n * NFEAT;
  float s = 0.f;
  #pragma unroll
  for (int k = 0; k < NFEAT; k++) s += xr[k] * W1[k * HID + j];
  out[g] = s;
}

// ---------------- GCN aggregation (gather form, no float atomics) -----------
// out[n] = dinv[n] * ( lin[n]*dinv[n] + sum_{s in N(n)} lin[s]*dinv[s] ) + bias

__global__ void k_agg(const float* __restrict__ lin, const float* __restrict__ dinv,
                      const int* __restrict__ rowptr, const int* __restrict__ csr,
                      const float* __restrict__ bias, float* outF, u16* outB, int relu) {
  __shared__ int   snb[256];
  __shared__ float sdv[256];
  int n = blockIdx.x;
  int t = threadIdx.x;
  float dn = dinv[n];
  float acc[4];
  #pragma unroll
  for (int q = 0; q < 4; q++) {
    int j = t + q * 256;
    acc[q] = (j < HID) ? lin[(size_t)n * HID + j] * dn : 0.0f;
  }
  int s0 = rowptr[n], s1 = rowptr[n + 1];
  for (int base = s0; base < s1; base += 256) {
    int cn = s1 - base; if (cn > 256) cn = 256;
    __syncthreads();
    if (t < cn) { int s = csr[base + t]; snb[t] = s; sdv[t] = dinv[s]; }
    __syncthreads();
    for (int i = 0; i < cn; i++) {
      int s = snb[i]; float dv = sdv[i];
      #pragma unroll
      for (int q = 0; q < 4; q++) {
        int j = t + q * 256;
        if (j < HID) acc[q] += lin[(size_t)s * HID + j] * dv;
      }
    }
  }
  #pragma unroll
  for (int q = 0; q < 4; q++) {
    int j = t + q * 256;
    if (j < HID) {
      float v = dn * acc[q] + bias[j];
      if (relu) v = v > 0.f ? v : 0.f;
      if (outF) outF[(size_t)n * HID + j] = v;
      if (outB) outB[(size_t)n * HID + j] = f2bf(v);
    }
  }
}

// ---------------- weight pre-pack into WMMA B-fragment layout ----------------
// element e of lane L in k-tile tk, col-tile j:  K = tk*32 + (L/16)*8 + (e%8) + (e/8)*16,
// N = j*16 + (L%16).  Stored contiguously: 16 bf16 per lane per fragment.

__global__ void k_pack(const float* __restrict__ W, u16* __restrict__ dst, int Kt) {
  int p = blockIdx.x * 256 + threadIdx.x;
  int total = NT16 * Kt * 512;
  if (p >= total) return;
  int e  = p & 15;
  int L  = (p >> 4) & 31;
  int tk = (p >> 9) % Kt;
  int j  = (p >> 9) / Kt;
  int half = L >> 4, nn = L & 15;
  int k = tk * 32 + half * 8 + (e & 7) + (e >> 3) * 16;
  dst[p] = f2bf(W[(size_t)k * HID + j * 16 + nn]);
}

// ---------------- WMMA bf16 GEMM: one wave = 32x64 output tile ---------------
// 2 A-fragments x 4 B-fragments -> 8 accumulators; 8 WMMAs per 12 b128 loads.
// Waves within a block share one column group (consecutive row tiles) so the
// packed-B stream hits in the WGP cache.

__global__ void k_gemm(const u16* __restrict__ A, const u16* __restrict__ Bp,
                       const float* __restrict__ bias,
                       float* outF, u16* outB,
                       int Mtiles32, int Kt, int lda, int ldo, int colOff) {
  int lane = threadIdx.x & 31;
  int wv   = threadIdx.x >> 5;
  int gw   = blockIdx.x * 8 + wv;
  int rowT = gw % Mtiles32;            // consecutive waves: consecutive row tiles
  int cg   = gw / Mtiles32;            // same column group within a block
  if (cg >= NCG) return;               // wave-uniform; EXEC stays all-ones

  v8f acc[2][4];
  #pragma unroll
  for (int m2 = 0; m2 < 2; m2++)
    #pragma unroll
    for (int c = 0; c < 4; c++)
      #pragma unroll
      for (int r = 0; r < 8; r++) acc[m2][c][r] = 0.0f;

  int half = lane >> 4, ln = lane & 15;
  // A fragment: lane (half,m) holds row m, K = {half*8+0..7, 16+half*8+0..7}
  const u16* abase0 = A + (size_t)(rowT * 32 + ln) * lda + half * 8;
  const u16* abase1 = abase0 + (size_t)16 * lda;
  size_t bbase = (size_t)(cg * 4) * Kt * 512 + (size_t)lane * 16;

  for (int t = 0; t < Kt; t++) {
    Frag16 a0, a1;
    const uint4* ap0 = (const uint4*)(abase0 + t * 32);
    const uint4* ap1 = (const uint4*)(abase1 + t * 32);
    a0.q[0] = ap0[0]; a0.q[1] = ap0[2];
    a1.q[0] = ap1[0]; a1.q[1] = ap1[2];
    #pragma unroll
    for (int c = 0; c < 4; c++) {
      Frag16 b;
      const uint4* bp = (const uint4*)(Bp + bbase + (size_t)c * Kt * 512 + (size_t)t * 512);
      b.q[0] = bp[0]; b.q[1] = bp[1];
      acc[0][c] = __builtin_amdgcn_wmma_f32_16x16x32_bf16(
          false, a0.v, false, b.v, (short)0, acc[0][c], false, false);
      acc[1][c] = __builtin_amdgcn_wmma_f32_16x16x32_bf16(
          false, a1.v, false, b.v, (short)0, acc[1][c], false, false);
    }
  }
  // D layout: lane = half*16 + n ; VGPR r holds M = half*8 + r
  #pragma unroll
  for (int m2 = 0; m2 < 2; m2++) {
    #pragma unroll
    for (int c = 0; c < 4; c++) {
      int col = (cg * 4 + c) * 16 + ln;
      float bv = bias ? bias[col] : 0.0f;
      #pragma unroll
      for (int r = 0; r < 8; r++) {
        int row = rowT * 32 + m2 * 16 + half * 8 + r;
        float v = acc[m2][c][r] + bv;
        if (outF) outF[(size_t)row * ldo + colOff + col] = v;
        else      outB[(size_t)row * ldo + colOff + col] = f2bf(v);
      }
    }
  }
}

// ---------------- pooling / concat ----------------

__global__ void k_gpool(const float* __restrict__ h2, float* __restrict__ gemb) {
  int b = blockIdx.x;
  for (int j = threadIdx.x; j < HID; j += 256) {
    float s = 0.f;
    for (int i = 0; i < NPM; i++) s += h2[(size_t)(b * NPM + i) * HID + j];
    gemb[(size_t)b * HID + j] = s * (1.0f / (float)NPM);
  }
}

__global__ void k_fgpool(const float* __restrict__ h2, const int* __restrict__ fg_ind,
                         const int* __restrict__ ptr, u16* __restrict__ pooled,
                         float* __restrict__ maskOut) {
  __shared__ int sidx[AMAX];
  __shared__ int scnt;
  int i = blockIdx.x, t = threadIdx.x;
  int b = i >> 5;
  if (t == 0) scnt = 0;
  __syncthreads();
  if (t < AMAX) {
    int v = fg_ind[i * AMAX + t];
    sidx[t] = v;
    if (v >= 0) atomicAdd(&scnt, 1);
  }
  __syncthreads();
  int cnt = scnt;
  float inv = 1.0f / (float)(cnt > 0 ? cnt : 1);
  int base = ptr[b];
  for (int j = t; j < HID; j += 256) {
    float s = 0.f;
    #pragma unroll
    for (int a = 0; a < AMAX; a++) {
      int v = sidx[a];
      if (v >= 0) s += h2[(size_t)(base + v) * HID + j];
    }
    pooled[(size_t)i * HID + j] = f2bf(s * inv);
  }
  if (t == 0) maskOut[i] = (cnt > 0) ? 1.0f : 0.0f;
}

// fill fused[:, 0:512]=fg_embed, [512:640]=pos_embed, [1600:2560]=global_emb
__global__ void k_fill(const int* __restrict__ fg_type, const float* __restrict__ fg_embed,
                       const float* __restrict__ pos_embed, const float* __restrict__ gemb,
                       u16* __restrict__ fused) {
  int i = blockIdx.x;
  int b = i >> 5, f = i & 31;
  int typ = fg_type[i];
  for (int c = threadIdx.x; c < 1600; c += 256) {
    int cc = (c < 640) ? c : c + HID;         // skip proj slice [640,1600)
    float v;
    if (c < FGE)             v = fg_embed[(size_t)typ * FGE + c];
    else if (c < FGE + POSE) v = pos_embed[f * POSE + (c - FGE)];
    else                     v = gemb[(size_t)b * HID + (c - 640)];
    fused[(size_t)i * FUSEIN + cc] = f2bf(v);
  }
}

// ---------------- launch ----------------

extern "C" void kernel_launch(void* const* d_in, const int* in_sizes, int n_in,
                              void* d_out, int out_size, void* d_ws, size_t ws_size,
                              hipStream_t stream) {
  const float* graph_x  = (const float*)d_in[0];
  const int*   edge     = (const int*)d_in[1];
  const int*   ptr      = (const int*)d_in[3];
  const int*   fg_type  = (const int*)d_in[4];
  const int*   fg_ind   = (const int*)d_in[5];
  const float* W1   = (const float*)d_in[6];
  const float* b1   = (const float*)d_in[7];
  const float* W2   = (const float*)d_in[8];
  const float* b2   = (const float*)d_in[9];
  const float* fgE  = (const float*)d_in[10];
  const float* posE = (const float*)d_in[11];
  const float* pW   = (const float*)d_in[12];
  const float* pb   = (const float*)d_in[13];
  const float* fW   = (const float*)d_in[14];
  const float* fb   = (const float*)d_in[15];

  const int* e_src = edge;
  const int* e_dst = edge + E_EDGES;

  char* ws = (char*)d_ws;
  size_t off = 0;
  auto carve = [&](size_t bytes) -> char* {
    char* p = ws + off;
    off = (off + bytes + 255) & ~(size_t)255;
    return p;
  };
  int*   deg    = (int*)  carve((size_t)N_NODES * 4);
  int*   rowptr = (int*)  carve((size_t)(N_NODES + 1) * 4);
  int*   cursor = (int*)  carve((size_t)N_NODES * 4);
  int*   csr    = (int*)  carve((size_t)E_EDGES * 4);
  float* dinv   = (float*)carve((size_t)N_NODES * 4);
  float* lin    = (float*)carve((size_t)N_NODES * HID * 4);   // lin1, then lin2, then reused as fused
  u16*   hbf    = (u16*)  carve((size_t)N_NODES * HID * 2);   // h1 bf16, later reused as pooled
  float* h2     = (float*)carve((size_t)N_NODES * HID * 4);
  float* gemb   = (float*)carve((size_t)B_MOL * HID * 4);
  u16*   w2p    = (u16*)  carve((size_t)HID * HID * 2);
  u16*   projp  = (u16*)  carve((size_t)HID * HID * 2);
  u16*   fusep  = (u16*)  carve((size_t)FUSEIN * HID * 2);
  // buffer reuse after GCN stage
  u16* pooled = hbf;           // 8192*960 bf16 <= N*HID bf16
  u16* fused  = (u16*)lin;     // 8192*2560 bf16 <= N*HID f32

  float* outFused = (float*)d_out;                       // [8192,960]
  float* outMask  = (float*)d_out + (size_t)BF * HID;    // [8192]

  // --- CSR build + dinv ---
  hipMemsetAsync(deg,    0, (size_t)N_NODES * 4, stream);
  hipMemsetAsync(cursor, 0, (size_t)N_NODES * 4, stream);
  k_count  <<<E_EDGES / 256, 256, 0, stream>>>(e_dst, deg);
  k_scan   <<<1, 1024, 0, stream>>>(deg, rowptr, dinv);
  k_scatter<<<E_EDGES / 256, 256, 0, stream>>>(e_src, e_dst, rowptr, cursor, csr);

  // --- weight packing (bf16 WMMA B-layout) ---
  k_pack<<<(NT16 * 30 * 512 + 255) / 256, 256, 0, stream>>>(W2, w2p, 30);
  k_pack<<<(NT16 * 30 * 512 + 255) / 256, 256, 0, stream>>>(pW, projp, 30);
  k_pack<<<(NT16 * 80 * 512 + 255) / 256, 256, 0, stream>>>(fW, fusep, 80);

  // --- GCN conv 1: lin1 = x@W1 ; aggregate + b1 + relu -> h1 (bf16) ---
  k_lin1<<<(N_NODES * HID) / 256, 256, 0, stream>>>(graph_x, W1, lin);
  k_agg <<<N_NODES, 256, 0, stream>>>(lin, dinv, rowptr, csr, b1, nullptr, hbf, 1);

  // --- GCN conv 2: lin2 = h1@W2 (WMMA) ; aggregate + b2 -> h2 (f32) ---
  k_gemm<<<(N_NODES / 32) * NCG / 8, 256, 0, stream>>>(
      hbf, w2p, nullptr, lin, nullptr, N_NODES / 32, 30, HID, HID, 0);
  k_agg <<<N_NODES, 256, 0, stream>>>(lin, dinv, rowptr, csr, b2, h2, nullptr, 0);

  // --- pooling ---
  k_gpool <<<B_MOL, 256, 0, stream>>>(h2, gemb);
  k_fgpool<<<BF, 256, 0, stream>>>(h2, fg_ind, ptr, pooled, outMask);

  // --- concat buffer: static/pos/global slices ---
  k_fill<<<BF, 256, 0, stream>>>(fg_type, fgE, posE, gemb, fused);

  // --- proj GEMM (WMMA): pooled@projW + projb -> fused[:, 640:1600] (bf16) ---
  k_gemm<<<(BF / 32) * NCG / 8, 256, 0, stream>>>(
      pooled, projp, pb, nullptr, fused, BF / 32, 30, HID, FUSEIN, 640);

  // --- fuse GEMM (WMMA): fused@fuseW + fuseb -> d_out (f32) ---
  k_gemm<<<(BF / 32) * NCG / 8, 256, 0, stream>>>(
      fused, fusep, fb, outFused, nullptr, BF / 32, 80, FUSEIN, HID, 0);

  (void)in_sizes; (void)n_in; (void)out_size; (void)ws_size;
}
